// Decoder_3770981286304
// MI455X (gfx1250) — compile-verified, matrix operand order checked by reading, not run on hardware
//
#include <hip/hip_runtime.h>

// Problem constants (from the reference)
#define TT   20
#define BB   4096
#define VV   1000
#define EE   256
#define HH   1024
#define KK   1280      // E + H (fused GEMM K)
#define SOS_TOK 1

// GEMM tiling
#define BM   128       // rows of batch per block
#define BN   32        // cols of H per block (x4 gates internally)
#define KC   32        // K chunk per WMMA
#define LDA  40        // padded LDS k-stride (bf16) -> conflict-free ds_load_b128
#define ABYTES (BM * LDA * 2)      // 10240 B per A buffer
#define BBYTES (4 * BN * LDA * 2)  // 10240 B per B buffer

typedef __attribute__((ext_vector_type(16))) __bf16 bf16x16;
typedef __attribute__((ext_vector_type(8)))  float  f32x8;

__device__ __forceinline__ bf16x16 frag_ld(const __bf16* p) {
    // 16-bit A/B fragment: lane holds 8 bf16 at kb..kb+7 and 8 at kb+16..kb+23
    union { uint4 u[2]; bf16x16 v; } f;
    f.u[0] = *(const uint4*)(p);
    f.u[1] = *(const uint4*)(p + 16);
    return f.v;
}

// CDNA5 async memory->LDS copy (ASYNCcnt path). Low 32 bits of a generic
// pointer to __shared__ are the wave-relative LDS byte offset (ISA 10.2).
__device__ __forceinline__ void async_copy_b128(void* lds_dst, const void* gsrc) {
    asm volatile("global_load_async_to_lds_b128 %0, %1, off"
                 :: "v"((unsigned)(unsigned long long)lds_dst),
                    "v"((unsigned long long)gsrc)
                 : "memory");
}
__device__ __forceinline__ void wait_async0() {
    asm volatile("s_wait_asynccnt 0x0" ::: "memory");
}

__device__ __forceinline__ float sigf(float x) { return 1.0f / (1.0f + __expf(-x)); }

// ---- one-time: Wcat = bf16([W_ih | W_hh]) row-major [4H, 1280]; bias = b_ih + b_hh
__global__ __launch_bounds__(256) void prep_weights(
    const float* __restrict__ W_ih, const float* __restrict__ W_hh,
    const float* __restrict__ b_ih, const float* __restrict__ b_hh,
    __bf16* __restrict__ Wcat, float* __restrict__ bias) {
    int id = blockIdx.x * 256 + threadIdx.x;       // [0, 4H*K/4)
    int n  = id / (KK / 4);
    int k  = (id % (KK / 4)) * 4;
    float4 v;
    if (k < EE) v = *(const float4*)&W_ih[(size_t)n * EE + k];
    else        v = *(const float4*)&W_hh[(size_t)n * HH + (k - EE)];
    union { __bf16 x[4]; uint2 u; } o;
    o.x[0] = (__bf16)v.x; o.x[1] = (__bf16)v.y; o.x[2] = (__bf16)v.z; o.x[3] = (__bf16)v.w;
    *(uint2*)&Wcat[(size_t)n * KK + k] = o.u;
    if (id < 4 * HH) bias[id] = b_ih[id] + b_hh[id];
}

// ---- one-time: XH0 h-half <- bf16(decoder_hidden); c <- 0
__global__ __launch_bounds__(256) void init_state(
    const float* __restrict__ h0, __bf16* __restrict__ XH0, float* __restrict__ cbuf) {
    int id = blockIdx.x * 256 + threadIdx.x;       // [0, B*H/4)
    int b  = id / (HH / 4);
    int j  = (id % (HH / 4)) * 4;
    float4 v = *(const float4*)&h0[(size_t)b * HH + j];
    union { __bf16 x[4]; uint2 u; } o;
    o.x[0] = (__bf16)v.x; o.x[1] = (__bf16)v.y; o.x[2] = (__bf16)v.z; o.x[3] = (__bf16)v.w;
    *(uint2*)&XH0[(size_t)b * KK + EE + j] = o.u;
    float4 z = {0.f, 0.f, 0.f, 0.f};
    *(float4*)&cbuf[(size_t)b * HH + j] = z;
}

// ---- per step: XH[:,0:E] <- bf16(emb[idx_t])
__global__ __launch_bounds__(256) void embed_gather(
    const int* __restrict__ labels, const float* __restrict__ emb,
    __bf16* __restrict__ XH, int t) {
    int id = blockIdx.x * 256 + threadIdx.x;       // [0, B*E/4)
    int b  = id / (EE / 4);
    int k  = (id % (EE / 4)) * 4;
    int row = (t == 0) ? SOS_TOK : labels[(size_t)b * TT + (t - 1)];
    float4 v = *(const float4*)&emb[(size_t)row * EE + k];
    union { __bf16 x[4]; uint2 u; } o;
    o.x[0] = (__bf16)v.x; o.x[1] = (__bf16)v.y; o.x[2] = (__bf16)v.z; o.x[3] = (__bf16)v.w;
    *(uint2*)&XH[(size_t)b * KK + k] = o.u;
}

// ---- per step: gates = XH @ Wcat^T + bias (4 gate slabs), LSTM cell,
//      h -> out + next XH (bf16), c updated in place.
//      Async (ASYNCcnt) double-buffered LDS staging; one barrier per K-chunk.
__global__ __launch_bounds__(256) void lstm_step(
    const __bf16* __restrict__ XH,  __bf16* __restrict__ XHn,
    float* __restrict__ cbuf,
    const __bf16* __restrict__ Wcat, const float* __restrict__ bias,
    float* __restrict__ out) {
    __shared__ __align__(16) char smem[2 * ABYTES + 2 * BBYTES];  // 40 KB

    const int tid  = threadIdx.x;
    const int wave = tid >> 5;
    const int lane = tid & 31;
    const int wm = wave >> 1;                  // 0..3  (32-row slab)
    const int wn = wave & 1;                   // 0..1  (16-col slab)
    const int blockM = blockIdx.x * BM;
    const int blockN = blockIdx.y * BN;

    const f32x8 zero = {0.f,0.f,0.f,0.f,0.f,0.f,0.f,0.f};
    f32x8 acc[2][4];                           // 2 M-subtiles x {i,f,g,o}
    #pragma unroll
    for (int s = 0; s < 2; ++s)
        #pragma unroll
        for (int g = 0; g < 4; ++g) acc[s][g] = zero;

    // --- per-thread async staging addresses (4x b128 per K-chunk) ---
    // A tile 128x32: chunk cid = tid -> row tid/4, and row+64 for cid+256
    const int r   = tid >> 2;
    const int ko  = (tid & 3) * 8;
    // B tiles 4x32x32: gates (tid>>7) and (tid>>7)+2 share col/koff
    const int g0   = tid >> 7;                 // 0..1
    const int rem  = tid & 127;
    const int colb = rem >> 2;
    const int kob  = (rem & 3) * 8;

    const __bf16* gA0 = XH   + (size_t)(blockM + r) * KK + ko;
    const __bf16* gA1 = gA0  + (size_t)64 * KK;
    const __bf16* gB0 = Wcat + (size_t)(g0 * HH + blockN + colb) * KK + kob;
    const __bf16* gB1 = gB0  + (size_t)2 * HH * KK;

    const int ldsA0 = (r * LDA + ko) * 2;
    const int ldsA1 = ldsA0 + 64 * LDA * 2;
    const int ldsB0 = ((g0 * BN + colb) * LDA + kob) * 2;
    const int ldsB1 = ldsB0 + 2 * BN * LDA * 2;

    auto stage = [&](int p, int kc) {
        char* a = smem + p * ABYTES;
        char* b = smem + 2 * ABYTES + p * BBYTES;
        async_copy_b128(a + ldsA0, gA0 + kc);
        async_copy_b128(a + ldsA1, gA1 + kc);
        async_copy_b128(b + ldsB0, gB0 + kc);
        async_copy_b128(b + ldsB1, gB1 + kc);
    };

    stage(0, 0);

    const int kb = (lane >> 4) * 8;            // ISA 16-bit fragment K striping
    const int fr = lane & 15;                  // fragment row/col within subtile

    for (int i = 0; i < KK / KC; ++i) {
        wait_async0();                         // my async writes to buf[i&1] done
        __syncthreads();                       // => everyone's writes done
        if (i + 1 < KK / KC) stage((i + 1) & 1, (i + 1) * KC);

        const __bf16* bufA = (const __bf16*)(smem + (i & 1) * ABYTES);
        const __bf16* bufB = (const __bf16*)(smem + 2 * ABYTES + (i & 1) * BBYTES);

        bf16x16 a0 = frag_ld(bufA + (wm * 32 +      fr) * LDA + kb);
        bf16x16 a1 = frag_ld(bufA + (wm * 32 + 16 + fr) * LDA + kb);
        bf16x16 bfr[4];
        #pragma unroll
        for (int g = 0; g < 4; ++g)
            bfr[g] = frag_ld(bufB + (g * BN + wn * 16 + fr) * LDA + kb);
        #pragma unroll
        for (int g = 0; g < 4; ++g) {
            acc[0][g] = __builtin_amdgcn_wmma_f32_16x16x32_bf16(
                false, a0, false, bfr[g], (short)0, acc[0][g], false, false);
            acc[1][g] = __builtin_amdgcn_wmma_f32_16x16x32_bf16(
                false, a1, false, bfr[g], (short)0, acc[1][g], false, false);
        }
        // single barrier per chunk: next iteration's top barrier orders
        // this iteration's LDS reads against next chunk's async writes
    }

    // epilogue: LSTM cell (C layout: VGPR j -> M = j + 8*(lane/16), N = lane%16)
    const int col_l    = lane & 15;
    const int row_base = (lane >> 4) * 8;
    const int cb = blockN + wn * 16 + col_l;
    const float bi = bias[cb];
    const float bf = bias[HH + cb];
    const float bg = bias[2 * HH + cb];
    const float bo = bias[3 * HH + cb];
    #pragma unroll
    for (int s = 0; s < 2; ++s) {
        #pragma unroll
        for (int j = 0; j < 8; ++j) {
            int rb = blockM + wm * 32 + s * 16 + row_base + j;
            float iv = sigf (acc[s][0][j] + bi);
            float fv = sigf (acc[s][1][j] + bf);
            float gv = tanhf(acc[s][2][j] + bg);
            float ov = sigf (acc[s][3][j] + bo);
            size_t ci = (size_t)rb * HH + cb;
            float cn = fv * cbuf[ci] + iv * gv;
            float hn = ov * tanhf(cn);
            cbuf[ci] = cn;
            out[ci]  = hn;
            XHn[(size_t)rb * KK + EE + cb] = (__bf16)hn;
        }
    }
}

extern "C" void kernel_launch(void* const* d_in, const int* in_sizes, int n_in,
                              void* d_out, int out_size, void* d_ws, size_t ws_size,
                              hipStream_t stream) {
    (void)in_sizes; (void)n_in; (void)out_size; (void)ws_size;
    const int*   labels = (const int*)  d_in[0];   // [B, T]
    const float* h0     = (const float*)d_in[1];   // [B, H]
    const float* emb    = (const float*)d_in[2];   // [V, E]
    const float* W_ih   = (const float*)d_in[3];   // [4H, E]
    const float* W_hh   = (const float*)d_in[4];   // [4H, H]
    const float* b_ih   = (const float*)d_in[5];   // [4H]
    const float* b_hh   = (const float*)d_in[6];   // [4H]
    float* out = (float*)d_out;                    // [T, B, H]

    // carve workspace (~48 MB total)
    size_t off = 0;
    auto carve = [&](size_t bytes) {
        void* p = (char*)d_ws + off;
        off += (bytes + 255) & ~(size_t)255;
        return p;
    };
    __bf16* Wcat = (__bf16*)carve((size_t)4 * HH * KK * sizeof(__bf16)); // 10 MB
    float*  bias = (float*) carve((size_t)4 * HH * sizeof(float));
    __bf16* XH0  = (__bf16*)carve((size_t)BB * KK * sizeof(__bf16));     // 10 MB
    __bf16* XH1  = (__bf16*)carve((size_t)BB * KK * sizeof(__bf16));     // 10 MB
    float*  cbuf = (float*) carve((size_t)BB * HH * sizeof(float));      // 16 MB

    prep_weights<<<(4 * HH * KK / 4) / 256, 256, 0, stream>>>(W_ih, W_hh, b_ih, b_hh, Wcat, bias);
    init_state<<<(BB * HH / 4) / 256, 256, 0, stream>>>(h0, XH0, cbuf);

    dim3 grid(BB / BM, HH / BN);   // 32 x 32 blocks
    for (int t = 0; t < TT; ++t) {
        __bf16* cur = (t & 1) ? XH1 : XH0;
        __bf16* nxt = (t & 1) ? XH0 : XH1;
        embed_gather<<<(BB * EE / 4) / 256, 256, 0, stream>>>(labels, emb, cur, t);
        lstm_step<<<grid, 256, 0, stream>>>(cur, nxt, cbuf, Wcat, bias,
                                            out + (size_t)t * BB * HH);
    }
}
